// PointNetWithFiLMDensity_57818849738974
// MI455X (gfx1250) — matrix operand order repaired and checked by reading, not compiled, not run on hardware
//
#include <hip/hip_runtime.h>
#include <hip/hip_bf16.h>
#include <math.h>

// ---------------------------------------------------------------------------
// CDNA5 (gfx1250) PointNet++ w/ FiLM.  Matrix work uses v_wmma_f32_16x16x32_f16
// (wave32, 16x16 tiles, f32 accum).  Layout assumptions follow CDNA5 ISA 7.12.2.
// All GEMM row counts are padded to multiples of the 64-row workgroup tile so
// the inner K-loop carries no per-lane guards (no exec-mask save/restore, no
// zero-fill VALU between WMMAs).
// ---------------------------------------------------------------------------

typedef _Float16 h16v __attribute__((ext_vector_type(16)));
typedef _Float16 h8v  __attribute__((ext_vector_type(8)));
typedef float    f8v  __attribute__((ext_vector_type(8)));

#define INF_F __builtin_inff()

union H16U { h16v v; h8v h[2]; _Float16 e[16]; };

__device__ __forceinline__ f8v wmma_f16(h16v a, h16v b, f8v c) {
  // 8 args: (neg_a, A, neg_b, B, c_mod, C, reuse_a, reuse_b)
  return __builtin_amdgcn_wmma_f32_16x16x32_f16(false, a, false, b, (short)0, c,
                                                false, false);
}

// A fragment (16x32 f16) from global row-major activations [rows x Kpad].
// ISA 7.12.2 16-bit A 16x32: lane L: M=L%16, g=L/16;
//   halfs 0-7  -> K = kk*32 + 8g + 0..7
//   halfs 8-15 -> K = kk*32 + 16 + 8g + 0..7
// Rows are padded to a multiple of 64 by construction -> unguarded loads.
__device__ __forceinline__ h16v load_a_global(const _Float16* __restrict__ A,
                                              int Kpad, int rowBase, int kk,
                                              int lane) {
  int m = rowBase + (lane & 15);
  int g = lane >> 4;
  const _Float16* p = A + (size_t)m * Kpad + kk * 32 + 8 * g;
  H16U u;
  u.h[0] = *(const h8v*)p;
  u.h[1] = *(const h8v*)(p + 16);
  return u.v;
}

// A fragment from a per-strip LDS region [16 x Npad] row-major.
__device__ __forceinline__ h16v load_a_lds(const _Float16* Hs, int Npad,
                                           int kk, int lane) {
  int m = lane & 15;
  int g = lane >> 4;
  const _Float16* p = Hs + m * Npad + kk * 32 + 8 * g;
  H16U u;
  u.h[0] = *(const h8v*)p;
  u.h[1] = *(const h8v*)(p + 16);
  return u.v;
}

// B fragment (32x16 f16).  Weights stored transposed [Npad x Kpad]; each
// lane's halfs 0-7 -> K = kk*32 + 8g + 0..7, halfs 8-15 -> K = kk*32 + 16 + 8g + 0..7, N = lane&15.
__device__ __forceinline__ h16v load_b_frag(const _Float16* __restrict__ Wt,
                                            int Kpad, int nt, int kk, int lane) {
  int n = nt * 16 + (lane & 15);
  const _Float16* p = Wt + (size_t)n * Kpad + kk * 32 + 8 * (lane >> 4);
  H16U u;
  u.h[0] = *(const h8v*)p;
  u.h[1] = *(const h8v*)(p + 16);
  return u.v;
}

// C/D layout (ISA 7.12.2): VGPR i -> M = i + 8*(lane/16), N = lane%16.
__device__ __forceinline__ void store_tile_lds(_Float16* Hs, int Npad, int nt,
                                               int lane, f8v c, float bv,
                                               bool relu) {
  int n = nt * 16 + (lane & 15);
  int g = lane >> 4;
#pragma unroll
  for (int i = 0; i < 8; ++i) {
    float v = c[i] + bv;
    if (relu) v = fmaxf(v, 0.0f);
    Hs[(i + 8 * g) * Npad + n] = (_Float16)v;
  }
}

// ---------------------------------------------------------------------------
// Fused 3-layer MLP.  blockDim = 128 (4 waves); each wave owns a 16-row strip,
// bounces layer activations through wave-private LDS (in-order DS ops, no
// cross-wave barriers needed until the final max-combine).
// Mode A (outRow != null): write rows x Nout f32 (guarded by rowsValid).
// Mode B (outMax != null): 64 rows per block = 1 center; masked max over the
//   64 neighbor rows -> outMax[center][Nout].
// rowsValid: actual (unpadded) row count; the A buffer itself always holds
// gridDim.x*64 readable rows.
// ---------------------------------------------------------------------------
__global__ void mlp3_wmma_kernel(
    const _Float16* __restrict__ A, int rowsValid, int Kpad,
    const _Float16* __restrict__ W0, const float* __restrict__ Bs0, int N1,
    const _Float16* __restrict__ W1, const float* __restrict__ Bs1, int N2,
    const _Float16* __restrict__ W2, const float* __restrict__ Bs2, int N3,
    int Nout,
    const int* __restrict__ nbrValid,
    float* __restrict__ outRow,
    float* __restrict__ outMax) {
  extern __shared__ char smem[];
  _Float16* H1 = (_Float16*)smem;           // 64 x N1
  _Float16* H2 = H1 + 64 * N1;              // 64 x N2
  float* Red = (float*)(H2 + 64 * N2);      // 4 x N3 (mode B)

  const int tid = threadIdx.x;
  const int lane = tid & 31;
  const int strip = tid >> 5;               // 0..3
  const int R0 = blockIdx.x * 64;
  const int rowBase = R0 + strip * 16;
  const int g = lane >> 4;

  _Float16* H1s = H1 + strip * 16 * N1;
  _Float16* H2s = H2 + strip * 16 * N2;

  // Layer 0: A(global f16) @ W0 + b0, ReLU -> H1 (LDS)
  for (int nt = 0; nt < (N1 >> 4); ++nt) {
    f8v c = {0.f, 0.f, 0.f, 0.f, 0.f, 0.f, 0.f, 0.f};
    for (int kk = 0; kk < (Kpad >> 5); ++kk) {
      h16v a = load_a_global(A, Kpad, rowBase, kk, lane);
      h16v b = load_b_frag(W0, Kpad, nt, kk, lane);
      c = wmma_f16(a, b, c);
    }
    store_tile_lds(H1s, N1, nt, lane, c, Bs0[nt * 16 + (lane & 15)], true);
  }
  // Layer 1: H1 @ W1 + b1, ReLU -> H2 (LDS)
  for (int nt = 0; nt < (N2 >> 4); ++nt) {
    f8v c = {0.f, 0.f, 0.f, 0.f, 0.f, 0.f, 0.f, 0.f};
    for (int kk = 0; kk < (N1 >> 5); ++kk) {
      h16v a = load_a_lds(H1s, N1, kk, lane);
      h16v b = load_b_frag(W1, N1, nt, kk, lane);
      c = wmma_f16(a, b, c);
    }
    store_tile_lds(H2s, N2, nt, lane, c, Bs1[nt * 16 + (lane & 15)], true);
  }
  // Layer 2: H2 @ W2 + b2 (no ReLU) -> outRow, or masked max -> outMax
  bool validR[8];
#pragma unroll
  for (int i = 0; i < 8; ++i) validR[i] = false;
  if (outMax) {
#pragma unroll
    for (int i = 0; i < 8; ++i) {
      int r = rowBase + i + 8 * g;
      validR[i] = (r < rowsValid) && (nbrValid == nullptr || nbrValid[r] >= 0);
    }
  }
  for (int nt = 0; nt < (N3 >> 4); ++nt) {
    f8v c = {0.f, 0.f, 0.f, 0.f, 0.f, 0.f, 0.f, 0.f};
    for (int kk = 0; kk < (N2 >> 5); ++kk) {
      h16v a = load_a_lds(H2s, N2, kk, lane);
      h16v b = load_b_frag(W2, N2, nt, kk, lane);
      c = wmma_f16(a, b, c);
    }
    float bv = Bs2[nt * 16 + (lane & 15)];
    if (outMax) {
      float m = -INF_F;
#pragma unroll
      for (int i = 0; i < 8; ++i) {
        float v = validR[i] ? (c[i] + bv) : -INF_F;
        m = fmaxf(m, v);
      }
      m = fmaxf(m, __shfl_xor(m, 16, 32));  // combine the two M-halves
      if (lane < 16) Red[strip * N3 + nt * 16 + lane] = m;
    } else {
#pragma unroll
      for (int i = 0; i < 8; ++i) {
        int r = rowBase + i + 8 * g;
        int n = nt * 16 + (lane & 15);
        if (r < rowsValid && n < Nout) outRow[(size_t)r * Nout + n] = c[i] + bv;
      }
    }
  }
  if (outMax) {
    __syncthreads();
    for (int col = tid; col < Nout; col += blockDim.x) {
      float m = fmaxf(fmaxf(Red[col], Red[N3 + col]),
                      fmaxf(Red[2 * N3 + col], Red[3 * N3 + col]));
      outMax[(size_t)blockIdx.x * Nout + col] = m;
    }
  }
}

// ---------------------------------------------------------------------------
// Weight conversion: f32 (a x b) row-major -> f16 transposed padded (Npad x Kpad)
// plus zero-padded f32 bias.
// ---------------------------------------------------------------------------
__global__ void convw_kernel(const float* __restrict__ W,
                             const float* __restrict__ bs, int a, int bdim,
                             int Kp, int Np, _Float16* __restrict__ Wt,
                             float* __restrict__ bd) {
  int idx = blockIdx.x * blockDim.x + threadIdx.x;
  int tot = Kp * Np;
  if (idx < tot) {
    int n = idx / Kp;
    int k = idx % Kp;
    float v = (k < a && n < bdim) ? W[(size_t)k * bdim + n] : 0.f;
    Wt[idx] = (_Float16)v;
  }
  if (idx < Np) bd[idx] = (idx < bdim) ? bs[idx] : 0.f;
}

// ---------------------------------------------------------------------------
// Farthest point sampling.  One block per batch (serial S-step scan, LDS
// argmax reduction with lowest-index tie-break to match jnp.argmax).
// Also emits gathered center positions.
// ---------------------------------------------------------------------------
__global__ void fps_kernel(const float* __restrict__ pos, int stride, int Np,
                           int S, int* __restrict__ idxOut,
                           float* __restrict__ ctrOut) {
  __shared__ float rv[256];
  __shared__ int rn[256];
  __shared__ float lastp[3];
  const int b = blockIdx.x;
  const int t = threadIdx.x;
  const float* P = pos + (size_t)b * Np * stride;
  float mind[8];
#pragma unroll
  for (int i = 0; i < 8; ++i) mind[i] = INF_F;
  int last = 0;
  for (int s = 0; s < S; ++s) {
    if (t == 0) {
      idxOut[b * S + s] = last;
      lastp[0] = P[(size_t)last * stride + 0];
      lastp[1] = P[(size_t)last * stride + 1];
      lastp[2] = P[(size_t)last * stride + 2];
      float* c = ctrOut + ((size_t)b * S + s) * 3;
      c[0] = lastp[0]; c[1] = lastp[1]; c[2] = lastp[2];
    }
    __syncthreads();
    float lx = lastp[0], ly = lastp[1], lz = lastp[2];
    float bv = -INF_F;
    int bn = 0;
#pragma unroll
    for (int i = 0; i < 8; ++i) {
      int n = i * 256 + t;
      if (n < Np) {
        float dx = P[(size_t)n * stride + 0] - lx;
        float dy = P[(size_t)n * stride + 1] - ly;
        float dz = P[(size_t)n * stride + 2] - lz;
        float d = dx * dx + dy * dy + dz * dz;
        float m = fminf(mind[i], d);
        mind[i] = m;
        if (m > bv) { bv = m; bn = n; }  // ascending n, strict > keeps lowest
      }
    }
    rv[t] = bv;
    rn[t] = bn;
    __syncthreads();
    for (int off = 128; off > 0; off >>= 1) {
      if (t < off) {
        float ov = rv[t + off];
        int on = rn[t + off];
        if (ov > rv[t] || (ov == rv[t] && on < rn[t])) { rv[t] = ov; rn[t] = on; }
      }
      __syncthreads();
    }
    last = rn[0];
  }
}

// ---------------------------------------------------------------------------
// Ball query: one wave per center; K rounds of wave-wide argmin over lane-
// private distances (lowest-index tie break) == K smallest d2 within radius.
// nbr = -1 marks invalid slots.
// ---------------------------------------------------------------------------
__global__ void ballq_kernel(const float* __restrict__ pos, int stride, int Np,
                             const float* __restrict__ ctr, int Sc, float r2,
                             int* __restrict__ nbr) {
  const int wid = threadIdx.x >> 5;
  const int lane = threadIdx.x & 31;
  const int c = blockIdx.x * 4 + wid;
  if (c >= 16 * Sc) return;
  const int b = c / Sc;
  const float* P = pos + (size_t)b * Np * stride;
  const float* C = ctr + (size_t)c * 3;
  const float cx = C[0], cy = C[1], cz = C[2];
  float dl[64];
  const int NI = (Np + 31) / 32;
  for (int i = 0; i < NI; ++i) {
    int n = i * 32 + lane;
    float d = INF_F;
    if (n < Np) {
      float dx = P[(size_t)n * stride + 0] - cx;
      float dy = P[(size_t)n * stride + 1] - cy;
      float dz = P[(size_t)n * stride + 2] - cz;
      float d2 = dx * dx + dy * dy + dz * dz;
      d = (d2 <= r2) ? d2 : INF_F;
    }
    dl[i] = d;
  }
  for (int k = 0; k < 64; ++k) {
    float bv = INF_F;
    int bn = 0x7fffffff;
    for (int i = 0; i < NI; ++i) {
      float v = dl[i];
      int n = i * 32 + lane;
      if (v < bv || (v == bv && n < bn)) { bv = v; bn = n; }
    }
    for (int off = 16; off > 0; off >>= 1) {
      float ov = __shfl_xor(bv, off, 32);
      int on = __shfl_xor(bn, off, 32);
      if (ov < bv || (ov == bv && on < bn)) { bv = ov; bn = on; }
    }
    if (lane == 0) nbr[(size_t)c * 64 + k] = (bv < INF_F) ? bn : -1;
    if (bv < INF_F && lane == (bn & 31)) dl[bn >> 5] = INF_F;
  }
}

// ---------------------------------------------------------------------------
// Gather/build padded f16 activation rows for each stage.
// ---------------------------------------------------------------------------
__global__ void build_a1_kernel(const float* __restrict__ data,
                                const int* __restrict__ nbr1,
                                const float* __restrict__ ctr1,
                                _Float16* __restrict__ A) {
  int r = blockIdx.x * blockDim.x + threadIdx.x;
  if (r >= 16 * 1024 * 64) return;
  int s = (r >> 6) & 1023;
  int b = r >> 16;
  int j = nbr1[r];
  _Float16* a = A + (size_t)r * 32;
  float v[7] = {0.f, 0.f, 0.f, 0.f, 0.f, 0.f, 0.f};
  if (j >= 0) {
    const float* p = data + ((size_t)b * 2048 + j) * 5;
    const float* c = ctr1 + ((size_t)((b << 10) + s)) * 3;
    v[0] = p[0]; v[1] = p[1]; v[2] = p[2]; v[3] = p[4];   // [xyz, dens]
    v[4] = p[0] - c[0]; v[5] = p[1] - c[1]; v[6] = p[2] - c[2];
  }
#pragma unroll
  for (int i = 0; i < 7; ++i) a[i] = (_Float16)v[i];
#pragma unroll
  for (int i = 7; i < 32; ++i) a[i] = (_Float16)0.f;
}

__global__ void build_a2_kernel(const float* __restrict__ x1,
                                const int* __restrict__ nbr2,
                                const float* __restrict__ ctr1,
                                const float* __restrict__ ctr2,
                                _Float16* __restrict__ A) {
  int r = blockIdx.x * blockDim.x + threadIdx.x;
  if (r >= 16 * 256 * 64) return;
  int s = (r >> 6) & 255;
  int b = r >> 14;
  int j = nbr2[r];
  _Float16* a = A + (size_t)r * 160;
  if (j >= 0) {
    const float* xs = x1 + ((size_t)((b << 10) + j)) * 128;
    for (int k = 0; k < 128; ++k) a[k] = (_Float16)xs[k];
    const float* pj = ctr1 + ((size_t)((b << 10) + j)) * 3;
    const float* cc = ctr2 + ((size_t)((b << 8) + s)) * 3;
    a[128] = (_Float16)(pj[0] - cc[0]);
    a[129] = (_Float16)(pj[1] - cc[1]);
    a[130] = (_Float16)(pj[2] - cc[2]);
    for (int k = 131; k < 160; ++k) a[k] = (_Float16)0.f;
  } else {
    for (int k = 0; k < 160; ++k) a[k] = (_Float16)0.f;
  }
}

__global__ void build_a3_kernel(const float* __restrict__ x2,
                                const float* __restrict__ ctr2,
                                _Float16* __restrict__ A) {
  int id = blockIdx.x * blockDim.x + threadIdx.x;
  if (id >= 4096 * 288) return;
  int r = id / 288;
  int k = id % 288;
  float v;
  if (k < 256) v = x2[(size_t)r * 256 + k];
  else if (k < 259) v = ctr2[(size_t)r * 3 + (k - 256)];
  else v = 0.f;
  A[id] = (_Float16)v;
}

// ---------------------------------------------------------------------------
// FiLM: x1 = (e @ Wg + bg) * x1 + (e @ Wb + bb), e = W_embed[zone] + b_embed
// ---------------------------------------------------------------------------
__global__ void film_kernel(const float* __restrict__ data,
                            const int* __restrict__ idx1,
                            const float* __restrict__ We,
                            const float* __restrict__ be,
                            const float* __restrict__ Wg,
                            const float* __restrict__ bg,
                            const float* __restrict__ Wb,
                            const float* __restrict__ bb,
                            float* __restrict__ x1) {
  int id = blockIdx.x * blockDim.x + threadIdx.x;
  if (id >= 16384 * 128) return;
  int row = id >> 7;
  int j = id & 127;
  int b = row >> 10;
  int s = row & 1023;
  int src = idx1[(b << 10) + s];
  int z = (int)data[((size_t)(b * 2048 + src)) * 5 + 3];
  float gam = bg[j], bet = bb[j];
#pragma unroll
  for (int i = 0; i < 16; ++i) {
    float e = We[z * 16 + i] + be[i];
    gam += e * Wg[i * 128 + j];
    bet += e * Wb[i * 128 + j];
  }
  x1[id] = gam * x1[id] + bet;
}

// Global max over the 256 sa3 rows per batch -> head input (f16), padded with
// zero rows 16..63 so the head GEMM needs no row guards.
__global__ void rowmax_kernel(const float* __restrict__ h3,
                              _Float16* __restrict__ gh) {
  int t = blockIdx.x * blockDim.x + threadIdx.x;
  if (t >= 64 * 1024) return;  // 64 padded rows x 1024
  int b = t >> 10;
  int c = t & 1023;
  if (b >= 16) {
    gh[t] = (_Float16)0.f;
    return;
  }
  float m = -INF_F;
  for (int s = 0; s < 256; ++s)
    m = fmaxf(m, h3[((size_t)((b << 8) + s)) * 1024 + c]);
  gh[t] = (_Float16)m;
}

// ---------------------------------------------------------------------------
// Host orchestration.  Input order = setup_inputs() insertion order:
// data, sa1(W,b x3), sa2(x3), sa3(x3), film_embed, film_gamma, film_beta,
// head(x3).
// ---------------------------------------------------------------------------
extern "C" void kernel_launch(void* const* d_in, const int* in_sizes, int n_in,
                              void* d_out, int out_size, void* d_ws,
                              size_t ws_size, hipStream_t stream) {
  (void)in_sizes; (void)n_in; (void)out_size; (void)ws_size;
  const float* data = (const float*)d_in[0];
  const float* Wsrc[12] = {
      (const float*)d_in[1],  (const float*)d_in[3],  (const float*)d_in[5],
      (const float*)d_in[7],  (const float*)d_in[9],  (const float*)d_in[11],
      (const float*)d_in[13], (const float*)d_in[15], (const float*)d_in[17],
      (const float*)d_in[25], (const float*)d_in[27], (const float*)d_in[29]};
  const float* Bsrc[12] = {
      (const float*)d_in[2],  (const float*)d_in[4],  (const float*)d_in[6],
      (const float*)d_in[8],  (const float*)d_in[10], (const float*)d_in[12],
      (const float*)d_in[14], (const float*)d_in[16], (const float*)d_in[18],
      (const float*)d_in[26], (const float*)d_in[28], (const float*)d_in[30]};
  const float* feW = (const float*)d_in[19];
  const float* feB = (const float*)d_in[20];
  const float* fgW = (const float*)d_in[21];
  const float* fgB = (const float*)d_in[22];
  const float* fbW = (const float*)d_in[23];
  const float* fbB = (const float*)d_in[24];

  char* base = (char*)d_ws;
  size_t off = 0;
  auto alloc = [&](size_t bytes) -> char* {
    char* p = base + off;
    off = (off + bytes + 255) & ~(size_t)255;
    return p;
  };

  struct LD { int a, b, Kp, Np; };
  const LD L[12] = {
      {7, 64, 32, 64},       {64, 64, 64, 64},     {64, 128, 64, 128},
      {131, 128, 160, 128},  {128, 128, 128, 128}, {128, 256, 128, 256},
      {259, 256, 288, 256},  {256, 512, 256, 512}, {512, 1024, 512, 1024},
      {1024, 512, 1024, 512},{512, 256, 512, 256}, {256, 49, 256, 64}};
  _Float16* Wt[12];
  float* Bp[12];
  for (int i = 0; i < 12; ++i) {
    Wt[i] = (_Float16*)alloc((size_t)L[i].Kp * L[i].Np * 2);
    Bp[i] = (float*)alloc((size_t)L[i].Np * 4);
  }
  int* idx1 = (int*)alloc((size_t)16 * 1024 * 4);
  float* ctr1 = (float*)alloc((size_t)16 * 1024 * 3 * 4);
  int* nbr1 = (int*)alloc((size_t)16 * 1024 * 64 * 4);
  int* idx2 = (int*)alloc((size_t)16 * 256 * 4);
  float* ctr2 = (float*)alloc((size_t)16 * 256 * 3 * 4);
  int* nbr2 = (int*)alloc((size_t)16 * 256 * 64 * 4);
  _Float16* Abuf = (_Float16*)alloc((size_t)262144 * 160 * 2);  // shared A1/A2/A3
  float* x1 = (float*)alloc((size_t)16384 * 128 * 4);
  float* x2 = (float*)alloc((size_t)4096 * 256 * 4);
  float* h3 = (float*)alloc((size_t)4096 * 1024 * 4);
  _Float16* gh = (_Float16*)alloc((size_t)64 * 1024 * 2);  // 64 padded rows

  // Weights -> f16 transposed + padded (L2-resident thereafter).
  for (int i = 0; i < 12; ++i) {
    int tot = L[i].Kp * L[i].Np;
    convw_kernel<<<(tot + 255) / 256, 256, 0, stream>>>(
        Wsrc[i], Bsrc[i], L[i].a, L[i].b, L[i].Kp, L[i].Np, Wt[i], Bp[i]);
  }

  // Stage 1
  fps_kernel<<<16, 256, 0, stream>>>(data, 5, 2048, 1024, idx1, ctr1);
  ballq_kernel<<<(16 * 1024) / 4, 128, 0, stream>>>(data, 5, 2048, ctr1, 1024,
                                                    0.2f * 0.2f, nbr1);
  build_a1_kernel<<<(1048576 + 255) / 256, 256, 0, stream>>>(data, nbr1, ctr1,
                                                             Abuf);
  {
    size_t smem = (size_t)(64 * 64 + 64 * 64) * 2 + (size_t)4 * 128 * 4;
    mlp3_wmma_kernel<<<16384, 128, smem, stream>>>(
        Abuf, 1048576, 32, Wt[0], Bp[0], 64, Wt[1], Bp[1], 64, Wt[2], Bp[2],
        128, 128, nbr1, nullptr, x1);
  }
  film_kernel<<<(16384 * 128) / 256, 256, 0, stream>>>(data, idx1, feW, feB,
                                                       fgW, fgB, fbW, fbB, x1);

  // Stage 2
  fps_kernel<<<16, 256, 0, stream>>>(ctr1, 3, 1024, 256, idx2, ctr2);
  ballq_kernel<<<(16 * 256) / 4, 128, 0, stream>>>(ctr1, 3, 1024, ctr2, 256,
                                                   0.4f * 0.4f, nbr2);
  build_a2_kernel<<<(262144 + 255) / 256, 256, 0, stream>>>(x1, nbr2, ctr1,
                                                            ctr2, Abuf);
  {
    size_t smem = (size_t)(64 * 128 + 64 * 128) * 2 + (size_t)4 * 256 * 4;
    mlp3_wmma_kernel<<<4096, 128, smem, stream>>>(
        Abuf, 262144, 160, Wt[3], Bp[3], 128, Wt[4], Bp[4], 128, Wt[5], Bp[5],
        256, 256, nbr2, nullptr, x2);
  }

  // sa3 + global max + head
  build_a3_kernel<<<((4096 * 288) + 255) / 256, 256, 0, stream>>>(x2, ctr2,
                                                                  Abuf);
  {
    size_t smem = (size_t)(64 * 256 + 64 * 512) * 2 + (size_t)4 * 1024 * 4;
    mlp3_wmma_kernel<<<64, 128, smem, stream>>>(
        Abuf, 4096, 288, Wt[6], Bp[6], 256, Wt[7], Bp[7], 512, Wt[8], Bp[8],
        1024, 1024, nullptr, h3, nullptr);
  }
  rowmax_kernel<<<((64 * 1024) + 255) / 256, 256, 0, stream>>>(h3, gh);
  {
    size_t smem = (size_t)(64 * 512 + 64 * 256) * 2 + (size_t)4 * 64 * 4;
    mlp3_wmma_kernel<<<1, 128, smem, stream>>>(
        gh, 16, 1024, Wt[9], Bp[9], 512, Wt[10], Bp[10], 256, Wt[11], Bp[11],
        64, 49, nullptr, (float*)d_out, nullptr);
  }
}
